// DiGressGNN_34213709480162
// MI455X (gfx1250) — matrix operand order repaired and checked by reading, not compiled, not run on hardware
//
#include <hip/hip_runtime.h>
#include <hip/hip_bf16.h>

// ---------------------------------------------------------------------------
// DiGress-style GNN for MI455X (gfx1250, wave32, WMMA 16x16x32 f16).
// - Heavy GEMMs (h@W2, node head, fused edge MLP) on v_wmma_f32_16x16x32_f16,
//   kc-outer / 16 independent accumulators per wave (no WMMA RAW chains).
// - B fragments staged block-wide through LDS with async global->LDS copies
//   (ASYNCcnt) double-buffered against WMMA compute.
// - Graph scatter (GCN propagation) uses f32 global atomics (L2-resident).
// ---------------------------------------------------------------------------

#define NNODE 100000
#define NEDGE 400000
#define NGRAPH 1024
#define NATOM 16
#define NBOND 5
#define HID 256

typedef __attribute__((ext_vector_type(16))) _Float16 v16h;
typedef __attribute__((ext_vector_type(8)))  float    v8f;
typedef __attribute__((ext_vector_type(4)))  unsigned int v4u;
typedef __attribute__((ext_vector_type(4)))  int      v4i;

union FragH { v4u u[2]; v16h h; };

#define AS1 __attribute__((address_space(1)))
#define AS3 __attribute__((address_space(3)))

// 16-byte async global->LDS copy (falls back to a synchronous vector copy).
__device__ __forceinline__ void cp_async16(_Float16* lds, const _Float16* g) {
#if __has_builtin(__builtin_amdgcn_global_load_async_to_lds_b128)
    __builtin_amdgcn_global_load_async_to_lds_b128(
        (AS1 v4i*)(void*)g, (AS3 v4i*)(void*)lds, 0, 0);
#else
    *(v4u*)lds = *(const v4u*)g;
#endif
}
__device__ __forceinline__ void wait_async_all() {
#if __has_builtin(__builtin_amdgcn_s_wait_asynccnt)
    __builtin_amdgcn_s_wait_asynccnt(0);
#endif
}

// A fragment (16x32 f16, row-major source, any row stride):
// lane L covers row m = L&15; halves 0..7 -> cols koff + (L>>4)*8 .. +7,
// halves 8..15 -> cols koff+16 + (L>>4)*8 .. +7  (CDNA5 ISA 7.12.2 layout).
__device__ __forceinline__ v16h load_a16(const _Float16* rowp, int koff, int lane) {
    int hs = (lane >> 4) & 1;
    FragH f;
    f.u[0] = *(const v4u*)(rowp + koff + hs * 8);
    f.u[1] = *(const v4u*)(rowp + koff + 16 + hs * 8);
    return f.h;
}

// B fragment from a packed fragment buffer (global or LDS): 16 contiguous halves.
__device__ __forceinline__ v16h load_b_pk(const _Float16* pk, int fragid, int lane) {
    const _Float16* p = pk + ((size_t)fragid * 32 + lane) * 16;
    FragH f;
    f.u[0] = *(const v4u*)(p);
    f.u[1] = *(const v4u*)(p + 8);
    return f.h;
}

// ---------------------------------------------------------------------------
// Degree / normalization
// ---------------------------------------------------------------------------
__global__ void k_init_deg(float* deg, int n) {
    int i = blockIdx.x * 256 + threadIdx.x;
    if (i < n) deg[i] = 1.0f;                    // self-loop
}
__global__ void k_deg_scatter(const int* __restrict__ dst, float* deg, int e) {
    int i = blockIdx.x * 256 + threadIdx.x;
    if (i < e) atomicAdd(&deg[dst[i]], 1.0f);
}
__global__ void k_dinv(const float* __restrict__ deg, float* dinv, int n) {
    int i = blockIdx.x * 256 + threadIdx.x;
    if (i < n) dinv[i] = rsqrtf(deg[i]);
}

// ---------------------------------------------------------------------------
// GCN1: xw = one_hot(x) @ W1   (W1 staged in LDS, 16 KB)
// ---------------------------------------------------------------------------
__global__ void k_gcn1_xw(const float* __restrict__ x, const float* __restrict__ W1,
                          float* __restrict__ xw) {
    __shared__ float sW[NATOM * HID];
    __shared__ float sx[NATOM];
    int i = blockIdx.x, c = threadIdx.x;
    #pragma unroll
    for (int r = 0; r < NATOM; ++r) sW[r * HID + c] = W1[r * HID + c];
    if (c < NATOM) sx[c] = x[(size_t)i * NATOM + c];
    __syncthreads();
    float acc = 0.f;
    #pragma unroll
    for (int a = 0; a < NATOM; ++a) acc += sx[a] * sW[a * HID + c];
    xw[(size_t)i * HID + c] = acc;
}

// ---------------------------------------------------------------------------
// GCN propagation: agg = xw * dinv^2 (self)  then  agg[dst] += xw[src]*coef
// ---------------------------------------------------------------------------
__global__ void k_self_msg(const float* __restrict__ xw, const float* __restrict__ dinv,
                           float* __restrict__ agg, long total) {
    long t = (long)blockIdx.x * 256 + threadIdx.x;
    if (t >= total) return;
    int i = (int)(t >> 8);
    float d = dinv[i];
    agg[t] = xw[t] * d * d;
}
__global__ void k_edge_scatter(const float* __restrict__ xw, const int* __restrict__ src,
                               const int* __restrict__ dst, const float* __restrict__ dinv,
                               float* __restrict__ agg, long total) {
    long t = (long)blockIdx.x * 256 + threadIdx.x;
    if (t >= total) return;
    int e = (int)(t >> 6);
    int c0 = (int)(t & 63) << 2;
    int s = src[e], d = dst[e];
    float coef = dinv[s] * dinv[d];
    const float4 xv = *(const float4*)(xw + (size_t)s * HID + c0);
    float* ap = agg + (size_t)d * HID + c0;
    atomicAdd(ap + 0, xv.x * coef);
    atomicAdd(ap + 1, xv.y * coef);
    atomicAdd(ap + 2, xv.z * coef);
    atomicAdd(ap + 3, xv.w * coef);
}

// relu(agg + b) -> f16
__global__ void k_relu_f16(const float* __restrict__ agg, const float* __restrict__ b,
                           _Float16* __restrict__ h16, long total) {
    long t = (long)blockIdx.x * 256 + threadIdx.x;
    if (t >= total) return;
    int c = (int)(t & (HID - 1));
    h16[t] = (_Float16)fmaxf(agg[t] + b[c], 0.f);
}

// h = relu(agg + b2) + temb[batch]  -> f16
__global__ void k_final_h(const float* __restrict__ agg, const float* __restrict__ b2,
                          const float* __restrict__ temb, const int* __restrict__ batch,
                          _Float16* __restrict__ h16, long total) {
    long t = (long)blockIdx.x * 256 + threadIdx.x;
    if (t >= total) return;
    int c = (int)(t & (HID - 1));
    int i = (int)(t >> 8);
    float v = fmaxf(agg[t] + b2[c], 0.f) + temb[(size_t)batch[i] * HID + c];
    h16[t] = (_Float16)v;
}

// ---------------------------------------------------------------------------
// Sinusoidal time embedding + Linear + SiLU:  temb[G, 256]
// ---------------------------------------------------------------------------
__global__ void k_time_embed(const int* __restrict__ tg, const float* __restrict__ tW,
                             const float* __restrict__ tb, float* __restrict__ temb) {
    int g = blockIdx.x, c = threadIdx.x;
    float t = (float)tg[g];
    float acc = tb[c];
    const float nl = -9.210340371976184f / 32.0f;   // -log(10000)/half
    for (int j = 0; j < 32; ++j) {
        float arg = t * __expf(nl * (float)j);
        acc += __sinf(arg) * tW[j * HID + c];
        acc += __cosf(arg) * tW[(32 + j) * HID + c];
    }
    temb[(size_t)g * HID + c] = acc / (1.f + __expf(-acc));  // SiLU
}

// ---------------------------------------------------------------------------
// Pack a row-major f32 [K, Nsrc] weight into per-lane WMMA B fragments (f16),
// zero-padding columns to a multiple of 16.
//   kcMajor=1: fragid = kc*ntiles + nt  (k-chunk contiguous, for LDS staging)
//   kcMajor=0: fragid = nt*kchunks + kc
// Half index within frag: k = kc*32 + (i>>3)*16 + (lane>>4)*8 + (i&7) ; n = nt*16 + (lane&15)
// ---------------------------------------------------------------------------
__global__ void k_pack_b(const float* __restrict__ src, _Float16* __restrict__ dst,
                         int K, int Nsrc, int ntiles, int kcMajor) {
    int kchunks = K >> 5;
    long total = (long)ntiles * kchunks * 512;
    long tid = (long)blockIdx.x * 256 + threadIdx.x;
    if (tid >= total) return;
    int i = (int)(tid & 15);
    int l = (int)((tid >> 4) & 31);
    int f = (int)(tid >> 9);
    int nt, kc;
    if (kcMajor) { kc = f / ntiles;  nt = f % ntiles; }
    else         { nt = f / kchunks; kc = f % kchunks; }
    int k = kc * 32 + ((i >> 3) << 4) + ((l >> 4) << 3) + (i & 7);
    int n = nt * 16 + (l & 15);
    float v = (n < Nsrc) ? src[(size_t)k * Nsrc + n] : 0.f;
    dst[tid] = (_Float16)v;
}

// issue one 16KB k-chunk stage (16 fragments) of a kc-major packed B into LDS
__device__ __forceinline__ void stage_b(const _Float16* __restrict__ pk, int kc,
                                        _Float16* ldst) {
    const _Float16* gsrc = pk + (size_t)kc * (16 * 512);
    int off = threadIdx.x * 8;                  // 8 halves = 16 bytes per copy
    cp_async16(ldst + off,        gsrc + off);
    cp_async16(ldst + off + 2048, gsrc + off + 2048);
    cp_async16(ldst + off + 4096, gsrc + off + 4096);
    cp_async16(ldst + off + 6144, gsrc + off + 6144);
}

// ---------------------------------------------------------------------------
// xw2 = h1 @ W2  : per wave a 16x256 row-strip (16 accumulators), B staged in
// LDS (double-buffered async), A fragment loaded once per k-chunk.
// ---------------------------------------------------------------------------
__global__ void __launch_bounds__(256) k_gemm_h_w2(const _Float16* __restrict__ A16,
                                                   const _Float16* __restrict__ pkB,
                                                   float* __restrict__ C, int Mtiles) {
    __shared__ _Float16 sB[2][16 * 512];        // 2 x 16KB
    int warp = threadIdx.x >> 5, lane = threadIdx.x & 31;
    int mt = blockIdx.x * 8 + warp;
    if (mt >= Mtiles) mt = Mtiles - 1;          // ragged tail: duplicate tile (benign)
    const _Float16* rowp = A16 + (size_t)(mt * 16 + (lane & 15)) * HID;

    stage_b(pkB, 0, sB[0]);
    wait_async_all();
    __syncthreads();

    v8f cm[16];
    #pragma unroll
    for (int nt = 0; nt < 16; ++nt) cm[nt] = v8f{};

    for (int kc = 0; kc < 8; ++kc) {
        if (kc < 7) stage_b(pkB, kc + 1, sB[(kc + 1) & 1]);
        v16h a = load_a16(rowp, kc * 32, lane);
        const _Float16* bb = sB[kc & 1];
        #pragma unroll
        for (int nt = 0; nt < 16; ++nt) {
            v16h b = load_b_pk(bb, nt, lane);
            cm[nt] = __builtin_amdgcn_wmma_f32_16x16x32_f16(false, a, false, b,
                                                            (short)0, cm[nt],
                                                            false, false);
        }
        if (kc < 7) { wait_async_all(); __syncthreads(); }
    }

    int n0 = lane & 15;
    int m0 = mt * 16 + ((lane >> 4) << 3);
    #pragma unroll
    for (int nt = 0; nt < 16; ++nt)
        #pragma unroll
        for (int v = 0; v < 8; ++v)
            C[(size_t)(m0 + v) * HID + nt * 16 + n0] = cm[nt][v];
}

// ---------------------------------------------------------------------------
// Generic small WMMA GEMM (node head): one 16x16 tile per wave.
// ---------------------------------------------------------------------------
__global__ void k_gemm_wmma(const _Float16* __restrict__ A16, const _Float16* __restrict__ pkB,
                            const float* __restrict__ bias, float* __restrict__ C,
                            int Mtiles, int Ntiles, int Kchunks, int Ncols) {
    int wid = blockIdx.x * (blockDim.x >> 5) + (threadIdx.x >> 5);
    if (wid >= Mtiles * Ntiles) return;
    int lane = threadIdx.x & 31;
    int mt = wid / Ntiles, nt = wid % Ntiles;
    const _Float16* rowp = A16 + (size_t)(mt * 16 + (lane & 15)) * (Kchunks * 32);
    v8f c = {};
    for (int kc = 0; kc < Kchunks; ++kc) {
        v16h a = load_a16(rowp, kc * 32, lane);
        v16h b = load_b_pk(pkB, nt * Kchunks + kc, lane);
        c = __builtin_amdgcn_wmma_f32_16x16x32_f16(false, a, false, b, (short)0, c,
                                                   false, false);
    }
    int n = nt * 16 + (lane & 15);
    float bv = bias ? bias[n] : 0.f;
    int m0 = mt * 16 + ((lane >> 4) << 3);
    #pragma unroll
    for (int v = 0; v < 8; ++v)
        C[(size_t)(m0 + v) * Ncols + n] = c[v] + bv;
}

// ---------------------------------------------------------------------------
// Fused edge MLP: per wave, 16 edges (exact tiling: 25000 tiles = 3125 x 8).
//   mid = silu([h[src] || h[dst]] @ eW1 + eb1)   -- 16 accumulators, kc-outer,
//                                                   eW1 staged in LDS (async)
//   out = mid @ eW2_pad + eb2                    -- C->A reshape via LDS tile
// ---------------------------------------------------------------------------
__global__ void __launch_bounds__(256) k_edge_mlp(
        const _Float16* __restrict__ h16,
        const int* __restrict__ srci, const int* __restrict__ dsti,
        const _Float16* __restrict__ pkE1, const _Float16* __restrict__ pkE2,
        const float* __restrict__ eb1, const float* __restrict__ eb2,
        float* __restrict__ out) {
    __shared__ _Float16 sB[2][16 * 512];        // 32KB: staged eW1 fragments
    __shared__ _Float16 smid[8][16 * 32];       // 8KB: per-wave silu reshape tile
    int warp = threadIdx.x >> 5, lane = threadIdx.x & 31;
    int t = blockIdx.x * 8 + warp;
    int m = lane & 15, hs = lane >> 4;
    int erow = t * 16 + m;
    const _Float16* srow = h16 + (size_t)srci[erow] * HID;
    const _Float16* drow = h16 + (size_t)dsti[erow] * HID;

    stage_b(pkE1, 0, sB[0]);
    wait_async_all();
    __syncthreads();

    v8f cm[16];
    #pragma unroll
    for (int nt = 0; nt < 16; ++nt) cm[nt] = v8f{};

    for (int kc = 0; kc < 16; ++kc) {           // K = 512 (src || dst)
        if (kc < 15) stage_b(pkE1, kc + 1, sB[(kc + 1) & 1]);
        const _Float16* rp = (kc < 8) ? srow : drow;
        v16h a = load_a16(rp, (kc & 7) * 32, lane);
        const _Float16* bb = sB[kc & 1];
        #pragma unroll
        for (int nt = 0; nt < 16; ++nt) {
            v16h b = load_b_pk(bb, nt, lane);
            cm[nt] = __builtin_amdgcn_wmma_f32_16x16x32_f16(false, a, false, b,
                                                            (short)0, cm[nt],
                                                            false, false);
        }
        if (kc < 15) { wait_async_all(); __syncthreads(); }
    }

    // second GEMM: silu'd 16x32 tiles -> A fragments via LDS, vs padded eW2
    _Float16* lds = smid[warp];
    v8f cl = {};
    for (int ntp = 0; ntp < 8; ++ntp) {
        #pragma unroll
        for (int sub = 0; sub < 2; ++sub) {
            int nt = ntp * 2 + sub;
            float bv = eb1[nt * 16 + m];
            #pragma unroll
            for (int v = 0; v < 8; ++v) {
                float x = cm[nt][v] + bv;
                float s = x / (1.f + __expf(-x));       // SiLU
                lds[(hs * 8 + v) * 32 + sub * 16 + m] = (_Float16)s;
            }
        }
        // DS ops are in-order per wave: wave-private store->load reshape.
        v16h a2 = load_a16(lds + m * 32, 0, lane);
        v16h b2 = load_b_pk(pkE2, ntp, lane);
        cl = __builtin_amdgcn_wmma_f32_16x16x32_f16(false, a2, false, b2,
                                                    (short)0, cl, false, false);
    }
    int n = lane & 15;
    if (n < NBOND) {
        float bv = eb2[n];
        int m0 = t * 16 + hs * 8;
        #pragma unroll
        for (int v = 0; v < 8; ++v)
            out[(size_t)(m0 + v) * NBOND + n] = cl[v] + bv;
    }
}

// ---------------------------------------------------------------------------
extern "C" void kernel_launch(void* const* d_in, const int* in_sizes, int n_in,
                              void* d_out, int out_size, void* d_ws, size_t ws_size,
                              hipStream_t stream) {
    (void)in_sizes; (void)n_in; (void)out_size; (void)ws_size;

    const float* x_oh  = (const float*)d_in[0];
    const int*   eidx  = (const int*)  d_in[1];
    const int*   tgr   = (const int*)  d_in[2];
    const int*   batch = (const int*)  d_in[3];
    const float* W1    = (const float*)d_in[4];
    const float* b1    = (const float*)d_in[5];
    const float* W2    = (const float*)d_in[6];
    const float* b2    = (const float*)d_in[7];
    const float* tW    = (const float*)d_in[8];
    const float* tb    = (const float*)d_in[9];
    const float* hW    = (const float*)d_in[10];
    const float* hb    = (const float*)d_in[11];
    const float* eW1   = (const float*)d_in[12];
    const float* eb1   = (const float*)d_in[13];
    const float* eW2   = (const float*)d_in[14];
    const float* eb2   = (const float*)d_in[15];

    const int* src = eidx;
    const int* dst = eidx + NEDGE;

    // workspace carve-out (256B aligned)
    char* ws = (char*)d_ws;
    size_t o = 0;
    auto carve = [&](size_t bytes) { size_t r = o; o += (bytes + 255) & ~(size_t)255; return r; };
    float*     deg   = (float*)(ws + carve((size_t)NNODE * 4));
    float*     dinv  = (float*)(ws + carve((size_t)NNODE * 4));
    float*     bufA  = (float*)(ws + carve((size_t)NNODE * HID * 4));   // xw / xw2
    float*     bufB  = (float*)(ws + carve((size_t)NNODE * HID * 4));   // agg
    _Float16*  h16   = (_Float16*)(ws + carve((size_t)NNODE * HID * 2));
    float*     temb  = (float*)(ws + carve((size_t)NGRAPH * HID * 4));
    _Float16*  pkW2  = (_Float16*)(ws + carve((size_t)HID * HID * 2));      // kc-major
    _Float16*  pkE1  = (_Float16*)(ws + carve((size_t)2 * HID * HID * 2));  // kc-major
    _Float16*  pkE2  = (_Float16*)(ws + carve((size_t)8 * 512 * 2));        // 1 n-tile (pad)
    _Float16*  pkHW  = (_Float16*)(ws + carve((size_t)8 * 512 * 2));        // 1 n-tile

    float* out_node = (float*)d_out;                          // [N, 16]
    float* out_edge = (float*)d_out + (size_t)NNODE * NATOM;  // [E, 5]

    const long elemsNH = (long)NNODE * HID;

    // 0) weight packing (tiny, independent)
    k_pack_b<<<(16 * 8  * 512 + 255) / 256, 256, 0, stream>>>(W2,  pkW2, HID,     HID,   16, 1);
    k_pack_b<<<(16 * 16 * 512 + 255) / 256, 256, 0, stream>>>(eW1, pkE1, 2 * HID, HID,   16, 1);
    k_pack_b<<<(8  * 512      + 255) / 256, 256, 0, stream>>>(eW2, pkE2, HID,     NBOND, 1,  0);
    k_pack_b<<<(8  * 512      + 255) / 256, 256, 0, stream>>>(hW,  pkHW, HID,     NATOM, 1,  0);

    // 1) degrees / symmetric normalization
    k_init_deg   <<<(NNODE + 255) / 256, 256, 0, stream>>>(deg, NNODE);
    k_deg_scatter<<<(NEDGE + 255) / 256, 256, 0, stream>>>(dst, deg, NEDGE);
    k_dinv       <<<(NNODE + 255) / 256, 256, 0, stream>>>(deg, dinv, NNODE);

    // 2) GCN layer 1
    k_gcn1_xw<<<NNODE, HID, 0, stream>>>(x_oh, W1, bufA);
    k_self_msg<<<(unsigned)((elemsNH + 255) / 256), 256, 0, stream>>>(bufA, dinv, bufB, elemsNH);
    k_edge_scatter<<<(unsigned)(((long)NEDGE * 64 + 255) / 256), 256, 0, stream>>>(
        bufA, src, dst, dinv, bufB, (long)NEDGE * 64);
    k_relu_f16<<<(unsigned)((elemsNH + 255) / 256), 256, 0, stream>>>(bufB, b1, h16, elemsNH);

    // 3) GCN layer 2: xw2 = h1 @ W2  (WMMA, async-staged B), then propagate
    k_gemm_h_w2<<<(NNODE / 16 + 7) / 8, 256, 0, stream>>>(h16, pkW2, bufA, NNODE / 16);
    k_self_msg<<<(unsigned)((elemsNH + 255) / 256), 256, 0, stream>>>(bufA, dinv, bufB, elemsNH);
    k_edge_scatter<<<(unsigned)(((long)NEDGE * 64 + 255) / 256), 256, 0, stream>>>(
        bufA, src, dst, dinv, bufB, (long)NEDGE * 64);

    // 4) time embedding, then h = relu(agg2 + b2) + temb[batch]
    k_time_embed<<<NGRAPH, HID, 0, stream>>>(tgr, tW, tb, temb);
    k_final_h<<<(unsigned)((elemsNH + 255) / 256), 256, 0, stream>>>(bufB, b2, temb, batch,
                                                                    h16, elemsNH);

    // 5) node head: logits = h @ hW + hb  (WMMA, Ncols = 16)
    k_gemm_wmma<<<(NNODE / 16 + 7) / 8, 256, 0, stream>>>(h16, pkHW, hb, out_node,
                                                          NNODE / 16, 1, HID / 32, NATOM);

    // 6) fused edge MLP (WMMA x2, async-staged eW1, SiLU via LDS reshape)
    k_edge_mlp<<<NEDGE / 16 / 8, 256, 0, stream>>>(h16, src, dst, pkE1, pkE2,
                                                   eb1, eb2, out_edge);
}
